// TopologicalAttention_53936199303453
// MI455X (gfx1250) — compile-verified, matrix operand order checked by reading, not run on hardware
//
#include <hip/hip_runtime.h>
#include <hip/hip_bf16.h>
#include <math.h>

typedef __attribute__((ext_vector_type(16))) _Float16 v16h;
typedef __attribute__((ext_vector_type(8)))  _Float16 v8h;
typedef __attribute__((ext_vector_type(2)))  _Float16 v2h;
typedef __attribute__((ext_vector_type(8)))  float    v8f;
typedef __attribute__((ext_vector_type(4)))  unsigned int u32x4;
typedef __attribute__((ext_vector_type(8)))  int      i32x8;
typedef __attribute__((ext_vector_type(4)))  int      i32x4;

#define B_   2
#define S_   2048
#define HID_ 1024
#define NH_  16
#define HD_  64
#define GRID_ 12
#define DECAY_ 0.3f

// ---------------------------------------------------------------------------
// WMMA helper (CDNA5 16x16x32 f16 -> f32), layouts per ISA 7.12.2
// ---------------------------------------------------------------------------
__device__ __forceinline__ v8f wmma_f16(v16h a, v16h b, v8f c) {
    return __builtin_amdgcn_wmma_f32_16x16x32_f16(
        /*neg_a=*/false, a, /*neg_b=*/false, b,
        /*c_mod=*/(short)0, c, /*reuse_a=*/false, /*reuse_b=*/false);
}

// A/B fragment loader: base -> element [row0][k0] of a row-major tile.
__device__ __forceinline__ v16h load_frag(const _Float16* base, int ld) {
    const int lane = threadIdx.x & 31;
    const int row  = lane & 15;
    const int khi  = (lane >> 4) << 3;
    v16h r;
#pragma unroll
    for (int v = 0; v < 8; ++v) {
        const int kb = ((v & 4) ? 16 : 0) + khi + ((v & 3) << 1);
        r[2 * v]     = base[row * ld + kb];
        r[2 * v + 1] = base[row * ld + kb + 1];
    }
    return r;
}

// Transposed B loader: tile stored [k][n]; base -> [0][col0].
__device__ __forceinline__ v16h load_frag_T(const _Float16* base, int ld) {
    const int lane = threadIdx.x & 31;
    const int col  = lane & 15;
    const int khi  = (lane >> 4) << 3;
    v16h r;
#pragma unroll
    for (int v = 0; v < 8; ++v) {
        const int kb = ((v & 4) ? 16 : 0) + khi + ((v & 3) << 1);
        r[2 * v]     = base[kb * ld + col];
        r[2 * v + 1] = base[(kb + 1) * ld + col];
    }
    return r;
}

// ---------------------------------------------------------------------------
// Tensor Data Mover: 2D f16 tile load Global -> LDS with row padding.
// D# bitfields per CDNA5 ISA.  6-arg builtin form
// (uint32x4 g0, int32x8 g1, int32x4, int32x4, int32x8, i32 cpol).
// ---------------------------------------------------------------------------
__device__ __forceinline__ void tdm_load_2d_f16(unsigned lds_byte_addr,
                                                const void* gaddr,
                                                unsigned tensor_d0,
                                                unsigned tensor_d1,
                                                unsigned tile_d0,
                                                unsigned tile_d1,
                                                unsigned stride0,
                                                unsigned pad_interval_code,
                                                unsigned pad_amount_code) {
    unsigned long long ga = (unsigned long long)(uintptr_t)gaddr;
    u32x4 g0 = {};
    g0[0] = 1u;                                   // count=1 (valid), user mode
    g0[1] = lds_byte_addr;                        // lds_addr
    g0[2] = (unsigned)ga;                         // global_addr[31:0]
    g0[3] = (unsigned)((ga >> 32) & 0x1FFFFFFu)   // global_addr[56:32]
            | (2u << 30);                         // type=2 (image)
    i32x8 g1 = {};
    g1[0] = (int)((1u << 16)                      // data_size = 2 bytes
                | (1u << 20)                      // pad_enable
                | (pad_interval_code << 22)
                | (pad_amount_code << 25));
    g1[1] = (int)((tensor_d0 & 0xFFFFu) << 16);   // tensor_dim0[15:0]
    g1[2] = (int)(((tensor_d0 >> 16) & 0xFFFFu)
                | ((tensor_d1 & 0xFFFFu) << 16)); // tensor_dim0 hi | dim1 lo
    g1[3] = (int)(((tensor_d1 >> 16) & 0xFFFFu)
                | ((tile_d0 & 0xFFFFu) << 16));   // dim1 hi | tile_dim0
    g1[4] = (int)(tile_d1 & 0xFFFFu);             // tile_dim1 | tile_dim2=0
    g1[5] = (int)stride0;                         // tensor_dim0_stride[31:0]
    g1[6] = 0;
    g1[7] = 0;
    i32x4 z4 = {};
    i32x8 z8 = {};
    __builtin_amdgcn_tensor_load_to_lds(g0, g1, z4, z4, z8, 0);
}

// ---------------------------------------------------------------------------
// Toroidal causal log-bias (analytic; matches reference)
// ---------------------------------------------------------------------------
__device__ __forceinline__ float toro_bias(int mq, int nk) {
    if (nk > mq) return -23.0258509299f; // log(1e-10)
    int xm = (mq / GRID_) % GRID_, ym = mq % GRID_;
    int xn = (nk / GRID_) % GRID_, yn = nk % GRID_;
    int dxa = xm - xn; if (dxa < 0) dxa = -dxa;
    int dya = ym - yn; if (dya < 0) dya = -dya;
    int dx = (dxa < GRID_ - dxa) ? dxa : GRID_ - dxa;
    int dy = (dya < GRID_ - dya) ? dya : GRID_ - dya;
    float dist = sqrtf((float)(dx * dx + dy * dy));
    return __logf(__expf(-DECAY_ * dist) + 1e-10f);
}

// ---------------------------------------------------------------------------
// Shared staging helpers for the projection GEMMs
// ---------------------------------------------------------------------------
#define P1_BM 128
#define P1_BN 64
#define P1_BK 32
#define P1_LDA (P1_BK + 8)  // 40 halves (80B rows, 16B aligned)

// Stage X tile (f32 -> f16): thread handles 16 contiguous cols of one row.
__device__ __forceinline__ void stage_A_f32(_Float16 (*As)[P1_LDA],
                                            const float* __restrict__ X,
                                            int m0, int k0, int tid) {
    const int row = tid >> 1;
    const int cb  = (tid & 1) * 16;
    const float* src = &X[(size_t)(m0 + row) * HID_ + k0 + cb];
    v8h h0, h1;
#pragma unroll
    for (int j = 0; j < 2; ++j) {
        float4 fv = ((const float4*)src)[j];
        h0[4 * j + 0] = (_Float16)fv.x; h0[4 * j + 1] = (_Float16)fv.y;
        h0[4 * j + 2] = (_Float16)fv.z; h0[4 * j + 3] = (_Float16)fv.w;
    }
#pragma unroll
    for (int j = 0; j < 2; ++j) {
        float4 fv = ((const float4*)src)[2 + j];
        h1[4 * j + 0] = (_Float16)fv.x; h1[4 * j + 1] = (_Float16)fv.y;
        h1[4 * j + 2] = (_Float16)fv.z; h1[4 * j + 3] = (_Float16)fv.w;
    }
    *(v8h*)&As[row][cb]     = h0;
    *(v8h*)&As[row][cb + 8] = h1;
}

// Stage W tile transposed ([n][k], f32 -> f16): thread handles a k-pair x 4 n.
__device__ __forceinline__ void stage_B_f32T(_Float16 (*Bs)[P1_LDA],
                                             const float* __restrict__ W,
                                             int n0, int k0, int tid) {
    const int kp = tid & 15;           // k pair index (k = 2*kp)
    const int nb = (tid >> 4) * 4;     // 4 consecutive n
    const float* r0 = &W[(size_t)(k0 + 2 * kp) * HID_ + n0 + nb];
    float4 a = *(const float4*)r0;
    float4 b = *(const float4*)(r0 + HID_);
    v2h p;
    p[0] = (_Float16)a.x; p[1] = (_Float16)b.x; *(v2h*)&Bs[nb + 0][2 * kp] = p;
    p[0] = (_Float16)a.y; p[1] = (_Float16)b.y; *(v2h*)&Bs[nb + 1][2 * kp] = p;
    p[0] = (_Float16)a.z; p[1] = (_Float16)b.z; *(v2h*)&Bs[nb + 2][2 * kp] = p;
    p[0] = (_Float16)a.w; p[1] = (_Float16)b.w; *(v2h*)&Bs[nb + 3][2 * kp] = p;
}

// ---------------------------------------------------------------------------
// Kernel 1: fused Q/K/V projection; f16 result in [B, NH, S, HD] layout.
// ---------------------------------------------------------------------------
__global__ __launch_bounds__(256)
void qkv_proj_kernel(const float* __restrict__ X,
                     const float* __restrict__ Wq, const float* __restrict__ bq,
                     const float* __restrict__ Wk, const float* __restrict__ bk,
                     const float* __restrict__ Wv, const float* __restrict__ bv,
                     _Float16* __restrict__ q16, _Float16* __restrict__ k16,
                     _Float16* __restrict__ v16) {
    const int z = blockIdx.z;
    const float* W    = (z == 0) ? Wq : (z == 1) ? Wk : Wv;
    const float* bias = (z == 0) ? bq : (z == 1) ? bk : bv;
    _Float16*   out   = (z == 0) ? q16 : (z == 1) ? k16 : v16;

    __shared__ _Float16 As[P1_BM][P1_LDA];
    __shared__ _Float16 Bs[P1_BN][P1_LDA];

    const int m0 = blockIdx.y * P1_BM;
    const int n0 = blockIdx.x * P1_BN;
    const int tid  = threadIdx.x;
    const int wave = tid >> 5;
    const int wm   = (wave >> 1) * 32;
    const int wn   = (wave & 1) * 32;

    v8f acc[2][2] = {};

    for (int k0 = 0; k0 < HID_; k0 += P1_BK) {
        stage_A_f32(As, X, m0, k0, tid);
        stage_B_f32T(Bs, W, n0, k0, tid);
        __syncthreads();

        v16h a0 = load_frag(&As[wm][0],      P1_LDA);
        v16h a1 = load_frag(&As[wm + 16][0], P1_LDA);
        v16h b0 = load_frag(&Bs[wn][0],      P1_LDA);
        v16h b1 = load_frag(&Bs[wn + 16][0], P1_LDA);
        acc[0][0] = wmma_f16(a0, b0, acc[0][0]);
        acc[0][1] = wmma_f16(a0, b1, acc[0][1]);
        acc[1][0] = wmma_f16(a1, b0, acc[1][0]);
        acc[1][1] = wmma_f16(a1, b1, acc[1][1]);
        __syncthreads();
    }

    const int lane = tid & 31;
    const int col  = lane & 15;
    const int hi   = lane >> 4;
#pragma unroll
    for (int i = 0; i < 2; ++i)
#pragma unroll
        for (int j = 0; j < 2; ++j)
#pragma unroll
            for (int r = 0; r < 8; ++r) {
                int m = m0 + wm + 16 * i + r + 8 * hi;
                int n = n0 + wn + 16 * j + col;
                float val = acc[i][j][r] + bias[n];
                int b = m >> 11, s = m & (S_ - 1);
                int h = n >> 6,  d = n & (HD_ - 1);
                out[(((size_t)(b * NH_ + h)) * S_ + s) * HD_ + d] = (_Float16)val;
            }
}

// ---------------------------------------------------------------------------
// Kernel 2: flash-style causal attention with toroidal bias.
// K/V tiles staged by the Tensor Data Mover with hardware row padding.
// ---------------------------------------------------------------------------
#define A_BQ 128
#define A_BK 32
#define A_LDK (HD_ + 8)   // 72 halves: TDM pads 4 DWORDs per 128B row
#define A_LDS (A_BK + 2)
#define A_LDP (A_BK + 8)

__global__ __launch_bounds__(256)
void attn_kernel(const _Float16* __restrict__ q16,
                 const _Float16* __restrict__ k16,
                 const _Float16* __restrict__ v16,
                 _Float16* __restrict__ attn16) {
    __shared__ _Float16 Ks[A_BK][A_LDK];
    __shared__ _Float16 Vs[A_BK][A_LDK];
    __shared__ float    Ss[A_BQ][A_LDS];
    __shared__ _Float16 Ps[A_BQ][A_LDP];
    __shared__ float    mI[A_BQ], lI[A_BQ], fS[A_BQ];

    const int bh   = blockIdx.y;
    const int b    = bh / NH_;
    const int h    = bh % NH_;
    const int q0   = blockIdx.x * A_BQ;
    const int tid  = threadIdx.x;
    const int wave = tid >> 5;
    const int lane = tid & 31;
    const int col  = lane & 15;
    const int hi   = lane >> 4;

    if (tid < A_BQ) { mI[tid] = -1e30f; lI[tid] = 0.0f; }

    const _Float16* qbase = q16 + ((size_t)bh * S_ + q0 + 16 * wave) * HD_;
    v16h qA[2];
    qA[0] = load_frag(qbase, HD_);
    qA[1] = load_frag(qbase + 32, HD_);

    const unsigned ldsK = (unsigned)(uintptr_t)&Ks[0][0];
    const unsigned ldsV = (unsigned)(uintptr_t)&Vs[0][0];

    v8f o[4] = {};
    __syncthreads();

    const int kend = q0 + A_BQ; // causal truncation
    for (int kk = 0; kk < kend; kk += A_BK) {
        // --- TDM: stage 32x64 f16 K and V tiles, padded rows (72h stride) ---
        if (wave == 0) {
            const void* kp = k16 + ((size_t)bh * S_ + kk) * HD_;
            const void* vp = v16 + ((size_t)bh * S_ + kk) * HD_;
            // rows: 64 halves = 32 DW -> pad_interval code 4; pad 4 DW -> code 3
            tdm_load_2d_f16(ldsK, kp, HD_, S_, HD_, A_BK, HD_, 4, 3);
            tdm_load_2d_f16(ldsV, vp, HD_, S_, HD_, A_BK, HD_, 4, 3);
            __builtin_amdgcn_s_wait_tensorcnt(0);
        }
        __syncthreads();

        // --- scores: S = Q K^T / 8 + bias ---
#pragma unroll
        for (int nt = 0; nt < 2; ++nt) {
            v8f sacc = {};
            sacc = wmma_f16(qA[0], load_frag(&Ks[nt * 16][0],  A_LDK), sacc);
            sacc = wmma_f16(qA[1], load_frag(&Ks[nt * 16][32], A_LDK), sacc);
#pragma unroll
            for (int r = 0; r < 8; ++r) {
                int ml = 16 * wave + r + 8 * hi;
                int nl = nt * 16 + col;
                Ss[ml][nl] = sacc[r] * 0.125f + toro_bias(q0 + ml, kk + nl);
            }
        }
        __syncthreads();

        // --- online softmax over this 32-key chunk ---
        {
            int row = tid >> 1;
            int cb  = (tid & 1) * 16;
            float mloc = -1e30f;
#pragma unroll
            for (int c = 0; c < 16; ++c) mloc = fmaxf(mloc, Ss[row][cb + c]);
            mloc = fmaxf(mloc, __shfl_xor(mloc, 1));
            float mprev = mI[row];
            float mnew  = fmaxf(mprev, mloc);
            float sum = 0.0f;
#pragma unroll
            for (int c = 0; c < 16; ++c) {
                float p = __expf(Ss[row][cb + c] - mnew);
                Ps[row][cb + c] = (_Float16)p;
                sum += p;
            }
            sum += __shfl_xor(sum, 1);
            if ((tid & 1) == 0) {
                fS[row] = __expf(mprev - mnew);
                lI[row] = lI[row] * fS[row] + sum;
                mI[row] = mnew;
            }
        }
        __syncthreads();

        // --- rescale accumulator, then O += P @ V ---
        v16h pA = load_frag(&Ps[16 * wave][0], A_LDP);
#pragma unroll
        for (int nt = 0; nt < 4; ++nt) {
#pragma unroll
            for (int r = 0; r < 8; ++r)
                o[nt][r] *= fS[16 * wave + r + 8 * hi];
            v16h bV = load_frag_T(&Vs[0][nt * 16], A_LDK);
            o[nt] = wmma_f16(pA, bV, o[nt]);
        }
        __syncthreads();
    }

    // --- normalize and write out in [B, S, HID] f16 layout ---
#pragma unroll
    for (int nt = 0; nt < 4; ++nt)
#pragma unroll
        for (int r = 0; r < 8; ++r) {
            int ml = 16 * wave + r + 8 * hi;
            float val = o[nt][r] / lI[ml];
            attn16[((size_t)b * S_ + q0 + ml) * HID_ + h * HD_ + nt * 16 + col] =
                (_Float16)val;
        }
}

// ---------------------------------------------------------------------------
// Kernel 3: output projection; A tile staged via TDM (already f16).
// ---------------------------------------------------------------------------
__global__ __launch_bounds__(256)
void out_proj_kernel(const _Float16* __restrict__ A16,
                     const float* __restrict__ Wo,
                     const float* __restrict__ bo,
                     float* __restrict__ out) {
    __shared__ _Float16 As[P1_BM][P1_LDA];
    __shared__ _Float16 Bs[P1_BN][P1_LDA];

    const int m0 = blockIdx.y * P1_BM;
    const int n0 = blockIdx.x * P1_BN;
    const int tid  = threadIdx.x;
    const int wave = tid >> 5;
    const int wm   = (wave >> 1) * 32;
    const int wn   = (wave & 1) * 32;

    const unsigned ldsA = (unsigned)(uintptr_t)&As[0][0];

    v8f acc[2][2] = {};

    for (int k0 = 0; k0 < HID_; k0 += P1_BK) {
        if (wave == 0) {
            // 128 rows x 32 halves; 64B rows = 16 DW -> interval code 3;
            // pad 4 DW -> code 3  => LDS stride 40 halves (= P1_LDA).
            tdm_load_2d_f16(ldsA, A16 + (size_t)m0 * HID_ + k0,
                            HID_, B_ * S_, P1_BK, P1_BM, HID_, 3, 3);
        }
        stage_B_f32T(Bs, Wo, n0, k0, tid);
        if (wave == 0) __builtin_amdgcn_s_wait_tensorcnt(0);
        __syncthreads();

        v16h a0 = load_frag(&As[wm][0],      P1_LDA);
        v16h a1 = load_frag(&As[wm + 16][0], P1_LDA);
        v16h b0 = load_frag(&Bs[wn][0],      P1_LDA);
        v16h b1 = load_frag(&Bs[wn + 16][0], P1_LDA);
        acc[0][0] = wmma_f16(a0, b0, acc[0][0]);
        acc[0][1] = wmma_f16(a0, b1, acc[0][1]);
        acc[1][0] = wmma_f16(a1, b0, acc[1][0]);
        acc[1][1] = wmma_f16(a1, b1, acc[1][1]);
        __syncthreads();
    }

    const int lane = tid & 31;
    const int col  = lane & 15;
    const int hi   = lane >> 4;
#pragma unroll
    for (int i = 0; i < 2; ++i)
#pragma unroll
        for (int j = 0; j < 2; ++j)
#pragma unroll
            for (int r = 0; r < 8; ++r) {
                int m = m0 + wm + 16 * i + r + 8 * hi;
                int n = n0 + wn + 16 * j + col;
                out[(size_t)m * HID_ + n] = acc[i][j][r] + bo[n];
            }
}

// ---------------------------------------------------------------------------
extern "C" void kernel_launch(void* const* d_in, const int* in_sizes, int n_in,
                              void* d_out, int out_size, void* d_ws, size_t ws_size,
                              hipStream_t stream) {
    const float* X  = (const float*)d_in[0];
    const float* Wq = (const float*)d_in[1];
    const float* bq = (const float*)d_in[2];
    const float* Wk = (const float*)d_in[3];
    const float* bk = (const float*)d_in[4];
    const float* Wv = (const float*)d_in[5];
    const float* bv = (const float*)d_in[6];
    const float* Wo = (const float*)d_in[7];
    const float* bo = (const float*)d_in[8];
    float* out = (float*)d_out;

    const size_t tokElems = (size_t)B_ * S_ * HID_;
    _Float16* q16 = (_Float16*)d_ws;
    _Float16* k16 = q16 + tokElems;
    _Float16* v16 = k16 + tokElems;
    _Float16* a16 = v16 + tokElems;

    dim3 blk(256);
    dim3 g1(HID_ / P1_BN, (B_ * S_) / P1_BM, 3);
    qkv_proj_kernel<<<g1, blk, 0, stream>>>(X, Wq, bq, Wk, bk, Wv, bv,
                                            q16, k16, v16);

    dim3 g2(S_ / A_BQ, B_ * NH_);
    attn_kernel<<<g2, blk, 0, stream>>>(q16, k16, v16, a16);

    dim3 g3(HID_ / P1_BN, (B_ * S_) / P1_BM);
    out_proj_kernel<<<g3, blk, 0, stream>>>(a16, Wo, bo, out);
}
